// Attention_33028298506971
// MI455X (gfx1250) — matrix operand order repaired and checked by reading, not compiled, not run on hardware
//
#include <hip/hip_runtime.h>

typedef __bf16 bf16;
typedef __attribute__((ext_vector_type(16))) __bf16 v16bf;
typedef __attribute__((ext_vector_type(8)))  float  v8f;
typedef __attribute__((ext_vector_type(4)))  unsigned int u32x4;
typedef __attribute__((ext_vector_type(8)))  int i32x8;
typedef __attribute__((ext_vector_type(4)))  int i32x4;

#define DIM   4096
#define NH    32
#define HD    128
#define BSZ   8
#define SEQ   512
#define STARTP 512
#define KVLEN 1024
#define MROWS (BSZ*SEQ)          // 4096
#define SCALE 0.08838834764831845f  // 1/sqrt(128)

__device__ __forceinline__ v8f wmma_bf16(v16bf a, v16bf b, v8f c) {
  return __builtin_amdgcn_wmma_f32_16x16x32_bf16(false, a, false, b, (short)0, c,
                                                 false, false);
}

__device__ __forceinline__ unsigned lds_off_u32(const void* p) {
  // low 32 bits of a flat shared-aperture address = LDS byte offset
  return (unsigned)(unsigned long long)(uintptr_t)p;
}

// Issue a TDM 2-D tile load: tile_d0 x tile_d1 elements of 2 bytes,
// row-major in memory with row stride `stride_elems`, packed row-major in LDS.
// D# bit layout per CDNA5 ISA ch.8 (group0: count/lds/global/type,
// group1: data_size, tensor dims, tile dims, dim0 stride).
// This toolchain exposes the 6-arg builtin: (g0, g1, g2, g3, g_extra, cpol).
__device__ __forceinline__ void tdm_load_2d_bf16(unsigned lds_off,
                                                 const void* gptr,
                                                 unsigned tensor_d0,
                                                 unsigned tensor_d1,
                                                 unsigned tile_d0,
                                                 unsigned tile_d1,
                                                 unsigned long long stride_el) {
  unsigned long long ga = (unsigned long long)(uintptr_t)gptr;
  u32x4 g0 = {};
  g0[0] = 1u;                                    // count=1 valid descriptor
  g0[1] = lds_off;                               // lds_addr
  g0[2] = (unsigned)(ga & 0xffffffffull);        // global_addr[31:0]
  g0[3] = (unsigned)((ga >> 32) & 0x1ffffffull)  // global_addr[56:32]
          | (2u << 30);                          // type=2 ("image")
  i32x8 g1 = {};
  unsigned long long d0 = tensor_d0, d1 = tensor_d1;
  g1[0] = (int)(1u << 16);                                   // data_size=1 -> 2B
  g1[1] = (int)((d0 & 0xffffull) << 16);                     // tensor_dim0[15:0]
  g1[2] = (int)(((d0 >> 16) & 0xffffull) | ((d1 & 0xffffull) << 16));
  g1[3] = (int)(((d1 >> 16) & 0xffffull) | ((unsigned long long)tile_d0 << 16));
  g1[4] = (int)(tile_d1 & 0xffffu);                          // tile_dim1 (dim2=0)
  g1[5] = (int)(stride_el & 0xffffffffull);                  // dim0_stride[31:0]
  g1[6] = (int)((stride_el >> 32) & 0xffffull);              // dim0_stride[47:32]
  g1[7] = 0;
  i32x4 g2 = {};
  i32x4 g3 = {};
  i32x8 g4 = {};
  __builtin_amdgcn_tensor_load_to_lds(g0, g1, g2, g3, g4, 0);
}

// ---------------------------------------------------------------- f32 -> bf16
__global__ __launch_bounds__(256) void cvt_f32_bf16(const float* __restrict__ s,
                                                    bf16* __restrict__ d, long n) {
  long i = (long)blockIdx.x * blockDim.x + threadIdx.x;
  if (i < n) d[i] = (bf16)s[i];
}

// -------------------------------------------------- C[M,N] = A[M,K] * B[N,K]^T
// A, B row-major bf16. Block tile 128x128, 256 threads (8 waves), k-step 32.
// LDS slabs are filled by the Tensor Data Mover, double-buffered, overlapped
// with WMMA compute (TENSORcnt pipelining).
template <typename OutT>
__global__ __launch_bounds__(256) void gemm_bf16_nt(const bf16* __restrict__ A,
                                                    const bf16* __restrict__ B,
                                                    OutT* __restrict__ C,
                                                    int M, int N, int Kd) {
  __shared__ bf16 lA[2][128 * 32];
  __shared__ bf16 lB[2][128 * 32];
  const int m0 = blockIdx.y * 128, n0 = blockIdx.x * 128;
  const int tid = threadIdx.x;
  const int wave = tid >> 5, lane = tid & 31;
  const int r = lane & 15, hi = lane >> 4;
  const int wm = wave >> 1, wn = wave & 1;   // 4x2 wave grid, wave tile 32x64

  v8f acc[2][4] = {};
  const int niter = Kd >> 5;

  if (wave == 0) {  // wave-uniform branch: single wave drives the TDM
    tdm_load_2d_bf16(lds_off_u32(lA[0]), A + (size_t)m0 * Kd,
                     (unsigned)Kd, (unsigned)(M - m0), 32, 128,
                     (unsigned long long)Kd);
    tdm_load_2d_bf16(lds_off_u32(lB[0]), B + (size_t)n0 * Kd,
                     (unsigned)Kd, (unsigned)(N - n0), 32, 128,
                     (unsigned long long)Kd);
  }

  for (int it = 0; it < niter; ++it) {
    const int k0 = it << 5;
    if (wave == 0) {
      if (it + 1 < niter) {  // kick DMA for next slab, then wait current pair
        tdm_load_2d_bf16(lds_off_u32(lA[(it + 1) & 1]),
                         A + (size_t)m0 * Kd + k0 + 32,
                         (unsigned)(Kd - k0 - 32), (unsigned)(M - m0), 32, 128,
                         (unsigned long long)Kd);
        tdm_load_2d_bf16(lds_off_u32(lB[(it + 1) & 1]),
                         B + (size_t)n0 * Kd + k0 + 32,
                         (unsigned)(Kd - k0 - 32), (unsigned)(N - n0), 32, 128,
                         (unsigned long long)Kd);
        __builtin_amdgcn_s_wait_tensorcnt(2);  // oldest (current) pair done
      } else {
        __builtin_amdgcn_s_wait_tensorcnt(0);
      }
    }
    __syncthreads();  // release consumers after TDM completion

    const bf16* tA = lA[it & 1];
    const bf16* tB = lB[it & 1];

    v16bf af[2], bfr[4];
#pragma unroll
    for (int mi = 0; mi < 2; ++mi) {
      int rb = wm * 32 + mi * 16 + r;
#pragma unroll
      for (int i = 0; i < 8; ++i) {
        int k = ((i >> 2) << 4) + (hi << 3) + ((i & 3) << 1);
        af[mi][2 * i]     = tA[rb * 32 + k];
        af[mi][2 * i + 1] = tA[rb * 32 + k + 1];
      }
    }
#pragma unroll
    for (int ni = 0; ni < 4; ++ni) {
      int cb = wn * 64 + ni * 16 + r;
#pragma unroll
      for (int i = 0; i < 8; ++i) {
        int k = ((i >> 2) << 4) + (hi << 3) + ((i & 3) << 1);
        bfr[ni][2 * i]     = tB[cb * 32 + k];
        bfr[ni][2 * i + 1] = tB[cb * 32 + k + 1];
      }
    }
#pragma unroll
    for (int mi = 0; mi < 2; ++mi)
#pragma unroll
      for (int ni = 0; ni < 4; ++ni)
        acc[mi][ni] = wmma_bf16(af[mi], bfr[ni], acc[mi][ni]);

    __syncthreads();  // all reads done before this buffer is DMA-refilled
  }

#pragma unroll
  for (int mi = 0; mi < 2; ++mi)
#pragma unroll
    for (int ni = 0; ni < 4; ++ni)
#pragma unroll
      for (int j = 0; j < 8; ++j) {
        int row = m0 + wm * 32 + mi * 16 + (hi << 3) + j;
        int col = n0 + wn * 64 + ni * 16 + r;
        C[(size_t)row * N + col] = static_cast<OutT>(acc[mi][ni][j]);
      }
}

// -------------------------------------------- RoPE q/k + scatter to [b,h,t,d]
__global__ __launch_bounds__(128) void rope_scatter(
    const bf16* __restrict__ qtmp, const bf16* __restrict__ ktmp,
    const bf16* __restrict__ vtmp, const int* __restrict__ pos_ids,
    bf16* __restrict__ qb, bf16* __restrict__ kb, bf16* __restrict__ vb) {
  int bhs = blockIdx.x;
  int s = bhs & (SEQ - 1);
  int h = (bhs >> 9) & (NH - 1);
  int b = bhs >> 14;
  int d = threadIdx.x;

  size_t src = ((size_t)(b * SEQ + s)) * DIM + h * HD + d;
  float qv = (float)qtmp[src];
  float kv = (float)ktmp[src];
  float qp = (d < 64) ? -(float)qtmp[src + 64] : (float)qtmp[src - 64];
  float kp = (d < 64) ? -(float)ktmp[src + 64] : (float)ktmp[src - 64];

  int pos = pos_ids[b * SEQ + s];
  int dp = d & 63;
  float inv = __powf(10000.f, -(float)(2 * dp) * (1.0f / (float)HD));
  float ang = (float)pos * inv;
  float c = __cosf(ang), sn = __sinf(ang);

  size_t qdst = (((size_t)(b * NH + h)) * SEQ + s) * HD + d;
  qb[qdst] = (bf16)(qv * c + qp * sn);
  int t = pos;  // START + s
  size_t kdst = (((size_t)(b * NH + h)) * KVLEN + t) * HD + d;
  kb[kdst] = (bf16)(kv * c + kp * sn);
  vb[kdst] = vtmp[src];
}

// ------------------------------ old cache entries (t < START) f32 -> bf16 kv
__global__ __launch_bounds__(128) void cache_cvt(const float* __restrict__ ck,
                                                 const float* __restrict__ cv,
                                                 bf16* __restrict__ kb,
                                                 bf16* __restrict__ vb) {
  int bht = blockIdx.x;
  int t = bht & (STARTP - 1);
  int h = (bht >> 9) & (NH - 1);
  int b = bht >> 14;
  int d = threadIdx.x;
  size_t src = (((size_t)(b * NH + h)) * 2048 + t) * HD + d;  // MAX_SEQ=2048
  size_t dst = (((size_t)(b * NH + h)) * KVLEN + t) * HD + d;
  kb[dst] = (bf16)ck[src];
  vb[dst] = (bf16)cv[src];
}

// ------------------------------------------- additive key-mask per (b, t)
__global__ __launch_bounds__(256) void build_maskadd(
    const unsigned char* __restrict__ cmask, const float* __restrict__ router,
    float* __restrict__ ma) {
  int i = blockIdx.x * blockDim.x + threadIdx.x;
  if (i >= BSZ * KVLEN) return;
  int b = i >> 10, t = i & (KVLEN - 1);
  float v;
  if (t < STARTP)
    v = cmask[b * 2048 + t] ? 0.f : -1e9f;
  else
    v = (router[((size_t)(b * SEQ + (t - STARTP))) * 2] != 0.f) ? 0.f : -1e9f;
  ma[i] = v;
}

// --------------------------------------------------------- flash attention
// grid (b*h, SEQ/64), 128 threads = 4 waves; each wave owns 16 q rows.
__global__ __launch_bounds__(128) void attn_fwd(
    const bf16* __restrict__ qb, const bf16* __restrict__ kb,
    const bf16* __restrict__ vb, const float* __restrict__ maskadd,
    bf16* __restrict__ attn) {
  const int bh = blockIdx.x;
  const int b = bh >> 5, h = bh & (NH - 1);
  const int stile = blockIdx.y;
  const int tid = threadIdx.x;
  const int wave = tid >> 5, lane = tid & 31;
  const int r = lane & 15, hi = lane >> 4;
  const int s_base = stile * 64 + wave * 16;

  const bf16* Qp = qb + (((size_t)(b * NH + h)) * SEQ + s_base) * HD;
  const bf16* Kp = kb + ((size_t)(b * NH + h)) * KVLEN * HD;
  const bf16* Vp = vb + ((size_t)(b * NH + h)) * KVLEN * HD;
  const float* map = maskadd + b * KVLEN;

  __shared__ bf16 lK[32 * HD];
  __shared__ bf16 lV[32 * HD];
  __shared__ bf16 lP[4][16 * 32];

  // Q fragments: 16 rows x 128 d = 4 A-frags, kept in registers
  v16bf qf[4];
#pragma unroll
  for (int kt = 0; kt < 4; ++kt)
#pragma unroll
    for (int i = 0; i < 8; ++i) {
      int k = ((i >> 2) << 4) + (hi << 3) + ((i & 3) << 1);
      qf[kt][2 * i]     = Qp[(size_t)r * HD + kt * 32 + k];
      qf[kt][2 * i + 1] = Qp[(size_t)r * HD + kt * 32 + k + 1];
    }

  v8f acc[8] = {};
  float runmax[8], runsum[8];
#pragma unroll
  for (int j = 0; j < 8; ++j) { runmax[j] = -3e38f; runsum[j] = 0.f; }

  int kv_hi = STARTP + stile * 64 + 64;           // last needed key + 1
  kv_hi = (kv_hi + 31) & ~31;
  if (kv_hi > KVLEN) kv_hi = KVLEN;

  for (int kv = 0; kv < kv_hi; kv += 32) {
    __syncthreads();
    // stage K,V chunk (32 x 128 bf16 each) into LDS: 512 uint4 / 128 threads
#pragma unroll
    for (int c = tid; c < 512; c += 128) {
      int t = c >> 4, off = (c & 15) * 8;
      *(uint4*)&lK[t * HD + off] = *(const uint4*)&Kp[(size_t)(kv + t) * HD + off];
      *(uint4*)&lV[t * HD + off] = *(const uint4*)&Vp[(size_t)(kv + t) * HD + off];
    }
    if (kv + 32 < kv_hi) {  // hint next chunk -> global_prefetch_b8
      __builtin_prefetch(&Kp[(size_t)(kv + 32 + (tid >> 2)) * HD], 0, 1);
      __builtin_prefetch(&Vp[(size_t)(kv + 32 + (tid >> 2)) * HD], 0, 1);
    }
    __syncthreads();

    // scores: 16 q rows x 32 kv cols  (two 16-col N tiles)
    v8f sc[2] = {};
#pragma unroll
    for (int ct = 0; ct < 2; ++ct)
#pragma unroll
      for (int kc = 0; kc < 4; ++kc) {
        v16bf bfr;
#pragma unroll
        for (int i = 0; i < 8; ++i) {
          int kk = ((i >> 2) << 4) + (hi << 3) + ((i & 3) << 1);
          bfr[2 * i]     = lK[(ct * 16 + r) * HD + kc * 32 + kk];
          bfr[2 * i + 1] = lK[(ct * 16 + r) * HD + kc * 32 + kk + 1];
        }
        sc[ct] = wmma_bf16(qf[kc], bfr, sc[ct]);
      }

    float ma0 = map[kv + r];
    float ma1 = map[kv + 16 + r];
    float val[2][8], factor[8];
#pragma unroll
    for (int j = 0; j < 8; ++j) {
      int qpos = STARTP + s_base + (hi << 3) + j;
      int t0 = kv + r, t1 = kv + 16 + r;
      val[0][j] = sc[0][j] * SCALE + ma0 + ((t0 <= qpos) ? 0.f : -1e9f);
      val[1][j] = sc[1][j] * SCALE + ma1 + ((t1 <= qpos) ? 0.f : -1e9f);
    }
    // online softmax per row (row = 8*hi + j; reductions stay in lane halves)
#pragma unroll
    for (int j = 0; j < 8; ++j) {
      float m = fmaxf(val[0][j], val[1][j]);
#pragma unroll
      for (int mk = 1; mk < 16; mk <<= 1) m = fmaxf(m, __shfl_xor(m, mk, 32));
      float nm = fmaxf(runmax[j], m);
      float fac = __expf(runmax[j] - nm);
      runmax[j] = nm;
      float p0 = __expf(val[0][j] - nm);
      float p1 = __expf(val[1][j] - nm);
      val[0][j] = p0;
      val[1][j] = p1;
      float ps = p0 + p1;
#pragma unroll
      for (int mk = 1; mk < 16; mk <<= 1) ps += __shfl_xor(ps, mk, 32);
      runsum[j] = runsum[j] * fac + ps;
      factor[j] = fac;
    }
#pragma unroll
    for (int dt = 0; dt < 8; ++dt)
#pragma unroll
      for (int j = 0; j < 8; ++j) acc[dt][j] *= factor[j];

    // stage probs (16x32) to per-wave LDS, reload as A fragment
#pragma unroll
    for (int ct = 0; ct < 2; ++ct)
#pragma unroll
      for (int j = 0; j < 8; ++j)
        lP[wave][((hi << 3) + j) * 32 + ct * 16 + r] = (bf16)val[ct][j];
    asm volatile("s_wait_dscnt 0" ::: "memory");  // intra-wave DS RAW
    v16bf pf;
#pragma unroll
    for (int i = 0; i < 8; ++i) {
      int k = ((i >> 2) << 4) + (hi << 3) + ((i & 3) << 1);
      pf[2 * i]     = lP[wave][r * 32 + k];
      pf[2 * i + 1] = lP[wave][r * 32 + k + 1];
    }
    // PV: out[16 x 128] += P[16 x 32] * V[32 x 128]
#pragma unroll
    for (int dt = 0; dt < 8; ++dt) {
      v16bf vf;
#pragma unroll
      for (int i = 0; i < 8; ++i) {
        int kk = ((i >> 2) << 4) + (hi << 3) + ((i & 3) << 1);
        vf[2 * i]     = lV[kk * HD + dt * 16 + r];
        vf[2 * i + 1] = lV[(kk + 1) * HD + dt * 16 + r];
      }
      acc[dt] = wmma_bf16(pf, vf, acc[dt]);
    }
  }

  // normalize + store to [b*SEQ + s, h*HD + d] bf16 (ready for wo GEMM)
#pragma unroll
  for (int dt = 0; dt < 8; ++dt)
#pragma unroll
    for (int j = 0; j < 8; ++j) {
      int row = s_base + (hi << 3) + j;
      float o = acc[dt][j] / runsum[j];
      attn[((size_t)(b * SEQ + row)) * DIM + h * HD + dt * 16 + r] = (bf16)o;
    }
}

// ----------------------------------------------------------------- launcher
extern "C" void kernel_launch(void* const* d_in, const int* in_sizes, int n_in,
                              void* d_out, int out_size, void* d_ws,
                              size_t ws_size, hipStream_t stream) {
  const float* x        = (const float*)d_in[0];
  const float* router   = (const float*)d_in[1];
  const float* cache_k  = (const float*)d_in[2];
  const float* cache_v  = (const float*)d_in[3];
  const unsigned char* cache_mask = (const unsigned char*)d_in[4];
  const float* wq       = (const float*)d_in[6];
  const float* wk       = (const float*)d_in[7];
  const float* wv       = (const float*)d_in[8];
  const float* wo       = (const float*)d_in[9];
  const int* position_ids = (const int*)d_in[10];

  // ---- workspace carve (bf16 buffers), 256B aligned
  char* p = (char*)d_ws;
  auto carve = [&](size_t bytes) {
    char* q = p;
    p += (bytes + 255) & ~(size_t)255;
    return q;
  };
  const size_t SQ = (size_t)MROWS * DIM;           // 16.7M elems
  bf16* xb   = (bf16*)carve(SQ * 2);
  bf16* wqb  = (bf16*)carve(SQ * 2);
  bf16* wkb  = (bf16*)carve(SQ * 2);
  bf16* wvb  = (bf16*)carve(SQ * 2);
  bf16* wob  = (bf16*)carve(SQ * 2);
  bf16* qtmp = (bf16*)carve(SQ * 2);
  bf16* ktmp = (bf16*)carve(SQ * 2);
  bf16* vtmp = (bf16*)carve(SQ * 2);
  bf16* qbuf = (bf16*)carve((size_t)BSZ * NH * SEQ * HD * 2);
  bf16* kbuf = (bf16*)carve((size_t)BSZ * NH * KVLEN * HD * 2);
  bf16* vbuf = (bf16*)carve((size_t)BSZ * NH * KVLEN * HD * 2);
  bf16* attn = (bf16*)carve(SQ * 2);
  float* ma  = (float*)carve((size_t)BSZ * KVLEN * 4);

  long n = (long)SQ;
  int cg = (int)((n + 255) / 256);
  cvt_f32_bf16<<<cg, 256, 0, stream>>>(x, xb, n);
  cvt_f32_bf16<<<cg, 256, 0, stream>>>(wq, wqb, n);
  cvt_f32_bf16<<<cg, 256, 0, stream>>>(wk, wkb, n);
  cvt_f32_bf16<<<cg, 256, 0, stream>>>(wv, wvb, n);
  cvt_f32_bf16<<<cg, 256, 0, stream>>>(wo, wob, n);

  dim3 gg(DIM / 128, MROWS / 128);
  gemm_bf16_nt<bf16><<<gg, 256, 0, stream>>>(xb, wqb, qtmp, MROWS, DIM, DIM);
  gemm_bf16_nt<bf16><<<gg, 256, 0, stream>>>(xb, wkb, ktmp, MROWS, DIM, DIM);
  gemm_bf16_nt<bf16><<<gg, 256, 0, stream>>>(xb, wvb, vtmp, MROWS, DIM, DIM);

  rope_scatter<<<BSZ * NH * SEQ, 128, 0, stream>>>(qtmp, ktmp, vtmp,
                                                   position_ids, qbuf, kbuf,
                                                   vbuf);
  cache_cvt<<<BSZ * NH * STARTP, 128, 0, stream>>>(cache_k, cache_v, kbuf,
                                                   vbuf);
  build_maskadd<<<(BSZ * KVLEN + 255) / 256, 256, 0, stream>>>(cache_mask,
                                                               router, ma);

  dim3 ag(BSZ * NH, SEQ / 64);
  attn_fwd<<<ag, 128, 0, stream>>>(qbuf, kbuf, vbuf, ma, attn);

  gemm_bf16_nt<float><<<gg, 256, 0, stream>>>(attn, wob, (float*)d_out, MROWS,
                                              DIM, DIM);
}
